// Net_32169305047372
// MI455X (gfx1250) — compile-verified, matrix operand order checked by reading, not run on hardware
//
#include <hip/hip_runtime.h>
#include <hip/hip_bf16.h>
#include <math.h>

typedef __attribute__((ext_vector_type(16))) _Float16 v16h;
typedef __attribute__((ext_vector_type(8)))  _Float16 v8h;
typedef __attribute__((ext_vector_type(8)))  float    v8f;

static constexpr int  BB  = 8;
static constexpr int  PP  = 4096;
static constexpr int  NNp = BB * PP;          // 32768
static constexpr int  SS1 = 2048;
static constexpr int  SS2 = 1024;
static constexpr int  KKn = 64;               // neighbors
static constexpr long E1  = (long)BB * SS1 * KKn;   // 1,048,576
static constexpr long E2  = (long)BB * SS2 * KKn;   //   524,288
static constexpr int  GSTAT = 256;            // stats partial blocks
static constexpr float EPSBN = 1e-5f;

// ---------------------------------------------------------------------------
// Farthest point sampling: one block (1024 thr) per cloud, S-1 serial steps.
// ---------------------------------------------------------------------------
__global__ void fps_kernel(const float* __restrict__ pos, int* __restrict__ idx, int S) {
    const int b = blockIdx.x;
    const float* pb = pos + (long)b * PP * 3;
    __shared__ float wv[32];
    __shared__ int   wi[32];
    __shared__ float sel[3];
    const int tid = threadIdx.x;           // 1024 threads, 4 pts each
    float mind[4];
    const float x0 = pb[0], y0 = pb[1], z0 = pb[2];
    #pragma unroll
    for (int j = 0; j < 4; ++j) {
        int p = tid + j * 1024;
        float dx = pb[p*3]-x0, dy = pb[p*3+1]-y0, dz = pb[p*3+2]-z0;
        mind[j] = dx*dx + dy*dy + dz*dz;
    }
    if (tid == 0) idx[b * S] = 0;
    for (int s = 1; s < S; ++s) {
        float bv = -1.f; int bi = 0;
        #pragma unroll
        for (int j = 0; j < 4; ++j) {
            int p = tid + j * 1024;
            if (mind[j] > bv) { bv = mind[j]; bi = p; }
        }
        #pragma unroll
        for (int off = 16; off > 0; off >>= 1) {     // wave32 argmax, tie->low idx
            float ov = __shfl_xor(bv, off, 32);
            int   oi = __shfl_xor(bi, off, 32);
            if (ov > bv || (ov == bv && oi < bi)) { bv = ov; bi = oi; }
        }
        const int lane = tid & 31, w = tid >> 5;
        if (lane == 0) { wv[w] = bv; wi[w] = bi; }
        __syncthreads();
        if (tid < 32) {
            bv = wv[tid]; bi = wi[tid];
            #pragma unroll
            for (int off = 16; off > 0; off >>= 1) {
                float ov = __shfl_xor(bv, off, 32);
                int   oi = __shfl_xor(bi, off, 32);
                if (ov > bv || (ov == bv && oi < bi)) { bv = ov; bi = oi; }
            }
            if (tid == 0) {
                idx[b * S + s] = bi;
                sel[0] = pb[bi*3]; sel[1] = pb[bi*3+1]; sel[2] = pb[bi*3+2];
            }
        }
        __syncthreads();
        const float sx = sel[0], sy = sel[1], sz = sel[2];
        #pragma unroll
        for (int j = 0; j < 4; ++j) {
            int p = tid + j * 1024;
            float dx = pb[p*3]-sx, dy = pb[p*3+1]-sy, dz = pb[p*3+2]-sz;
            float d = dx*dx + dy*dy + dz*dz;
            mind[j] = fminf(mind[j], d);
        }
        __syncthreads();
    }
}

// ---------------------------------------------------------------------------
// K nearest within radius: 2 waves/block, one center per wave; d2 in LDS.
// ---------------------------------------------------------------------------
__global__ void knn_kernel(const float* __restrict__ pos, const int* __restrict__ idx,
                           int S, float r2, int* __restrict__ nbr, int* __restrict__ valid) {
    __shared__ float sD[2 * PP];                       // 32 KB
    const int tid = threadIdx.x, wave = tid >> 5, lane = tid & 31;
    const long center = (long)blockIdx.x * 2 + wave;   // < B*S always (exact grids)
    const int b = (int)(center / S), s = (int)(center % S);
    const float* pb = pos + (long)b * PP * 3;
    const int ci = idx[b * S + s];
    const float cx = pb[ci*3], cy = pb[ci*3+1], cz = pb[ci*3+2];
    float* D = &sD[wave * PP];
    for (int p = lane; p < PP; p += 32) {
        float dx = pb[p*3]-cx, dy = pb[p*3+1]-cy, dz = pb[p*3+2]-cz;
        D[p] = dx*dx + dy*dy + dz*dz;
    }
    __syncthreads();
    for (int k = 0; k < KKn; ++k) {
        float bv = 3.4e38f; int bi = 0x7fffffff;
        for (int p = lane; p < PP; p += 32) {
            float v = D[p];
            if (v < bv || (v == bv && p < bi)) { bv = v; bi = p; }
        }
        #pragma unroll
        for (int off = 16; off > 0; off >>= 1) {
            float ov = __shfl_xor(bv, off, 32);
            int   oi = __shfl_xor(bi, off, 32);
            if (ov < bv || (ov == bv && oi < bi)) { bv = ov; bi = oi; }
        }
        if (lane == 0) {
            nbr[center * KKn + k]   = bi;
            valid[center * KKn + k] = (bv <= r2) ? 1 : 0;
            D[bi] = 3.4e38f;
        }
        __syncthreads();
    }
}

// ---------------------------------------------------------------------------
// Gather kernels: build padded f32 feature rows (pad channels zeroed).
// ---------------------------------------------------------------------------
__global__ void gather1_kernel(const float* __restrict__ pos, const int* __restrict__ nbr,
                               float* __restrict__ Hin) {          // [E1, 32]
    const long total = E1 * 32;
    for (long i = (long)blockIdx.x * blockDim.x + threadIdx.x; i < total;
         i += (long)gridDim.x * blockDim.x) {
        long e = i >> 5; int c = (int)(i & 31);
        float v = 0.f;
        if (c < 3) {
            int b = (int)(e / ((long)SS1 * KKn));
            int s = (int)((e / KKn) % SS1);
            int j = nbr[e];
            float pj = pos[((long)b * PP + j) * 3 + c];
            float pi = pos[((long)b * SS1 + s) * 3 + c];   // faithful quirk: row b*S+s
            v = pj - pi;
        }
        Hin[i] = v;
    }
}

__global__ void gather2_kernel(const float* __restrict__ pos, const int* __restrict__ nbr,
                               const float* __restrict__ x1, float* __restrict__ Hin) { // [E2,64]
    const long total = E2 * 64;
    for (long i = (long)blockIdx.x * blockDim.x + threadIdx.x; i < total;
         i += (long)gridDim.x * blockDim.x) {
        long e = i >> 6; int c = (int)(i & 63);
        float v = 0.f;
        int b = (int)(e / ((long)SS2 * KKn));
        int s = (int)((e / KKn) % SS2);
        if (c < 32) {
            int j = nbr[e];
            v = x1[((long)b * PP + j) * 32 + c];
        } else if (c < 35) {
            int j = nbr[e]; int cc = c - 32;
            float pj = pos[((long)b * PP + j) * 3 + cc];
            float pi = pos[((long)b * SS2 + s) * 3 + cc];
            v = pj - pi;
        }
        Hin[i] = v;
    }
}

__global__ void gather3_kernel(const float* __restrict__ x2, const float* __restrict__ pos,
                               float* __restrict__ Hin) {          // [N, 96]
    const long total = (long)NNp * 96;
    for (long i = (long)blockIdx.x * blockDim.x + threadIdx.x; i < total;
         i += (long)gridDim.x * blockDim.x) {
        long row = i / 96; int c = (int)(i % 96);
        float v = 0.f;
        if (c < 64)       v = x2[row * 64 + c];
        else if (c < 67)  v = pos[row * 3 + (c - 64)];
        Hin[i] = v;
    }
}

// ---------------------------------------------------------------------------
// Weight prep: W f32[cin,cout] -> Wt f16[cout][KPAD], zero-padded K.
// ---------------------------------------------------------------------------
__global__ void prep_w_kernel(const float* __restrict__ W, int cin, int cout, int KPAD,
                              _Float16* __restrict__ Wt) {
    const long total = (long)cout * KPAD;
    for (long i = (long)blockIdx.x * blockDim.x + threadIdx.x; i < total;
         i += (long)gridDim.x * blockDim.x) {
        int n = (int)(i / KPAD), k = (int)(i % KPAD);
        Wt[i] = (k < cin) ? (_Float16)W[(long)k * cout + n] : (_Float16)0.f;
    }
}

// ---------------------------------------------------------------------------
// Generic WMMA GEMM: out[E,outStride] = act(in[E,KPAD]) @ W + bias
//   act = identity (ss==null)  or  relu(x*scale+shift) (fused BN)
// Block = 256 thr (8 waves); each wave owns a 16-row tile; grid.y tiles cout.
// ---------------------------------------------------------------------------
template <int KPAD, int CTILE>
__global__ void gemm_kernel(const float* __restrict__ in, const _Float16* __restrict__ Wt,
                            const float* __restrict__ bias, const float* __restrict__ ss,
                            float* __restrict__ out, int outStride) {
    __shared__ _Float16 sWt[CTILE * KPAD];
    __shared__ _Float16 sA[8 * 16 * KPAD];
    __shared__ float sScale[KPAD], sShift[KPAD];

    const int tid = threadIdx.x, wave = tid >> 5, lane = tid & 31;
    const long rowBase = (long)blockIdx.x * 128;
    const int  coutBase = blockIdx.y * CTILE;

    if (ss) {
        for (int i = tid; i < KPAD; i += 256) { sScale[i] = ss[i]; sShift[i] = ss[KPAD + i]; }
    }
    for (int i = tid; i < CTILE * KPAD; i += 256)
        sWt[i] = Wt[(long)coutBase * KPAD + i];
    __syncthreads();

    { // stage this wave's 16 rows (BN+ReLU fused), f32 -> f16
        const long base = (rowBase + (long)wave * 16) * KPAD;
        _Float16* aw = &sA[wave * 16 * KPAD];
        for (int i = lane; i < 16 * KPAD; i += 32) {
            float v = in[base + i];
            if (ss) { int k = i % KPAD; v = fmaxf(v * sScale[k] + sShift[k], 0.f); }
            aw[i] = (_Float16)v;
        }
    }
    __syncthreads();

    const int g = lane >> 4;           // half-wave group: K offset 0 / 8
    const int hr = lane & 15;          // row (A) / col (B) within tile
    const _Float16* aw = &sA[wave * 16 * KPAD];

    for (int nt = 0; nt < CTILE / 16; ++nt) {
        v8f acc = {0.f,0.f,0.f,0.f,0.f,0.f,0.f,0.f};
        #pragma unroll
        for (int kc = 0; kc < KPAD / 32; ++kc) {
            const int k0 = kc * 32 + g * 8;
            // A operand: row hr, halves k0..k0+7 and k0+16..k0+23 (ISA 16-bit A layout)
            const v8h alo = *reinterpret_cast<const v8h*>(aw + hr * KPAD + k0);
            const v8h ahi = *reinterpret_cast<const v8h*>(aw + hr * KPAD + k0 + 16);
            // B operand: Wt row = output col (nt*16+hr), same K striping
            const v8h blo = *reinterpret_cast<const v8h*>(&sWt[(nt * 16 + hr) * KPAD + k0]);
            const v8h bhi = *reinterpret_cast<const v8h*>(&sWt[(nt * 16 + hr) * KPAD + k0 + 16]);
            v16h a, bm;
            #pragma unroll
            for (int i = 0; i < 8; ++i) { a[i] = alo[i]; a[i+8] = ahi[i]; bm[i] = blo[i]; bm[i+8] = bhi[i]; }
            acc = __builtin_amdgcn_wmma_f32_16x16x32_f16(false, a, false, bm,
                                                         (short)0, acc, false, false);
        }
        #pragma unroll
        for (int r = 0; r < 8; ++r) {          // D layout: lanes0-15 M=r, lanes16-31 M=r+8
            const int m = r + 8 * g;
            const long row = rowBase + (long)wave * 16 + m;
            const int c = coutBase + nt * 16 + hr;
            out[row * outStride + c] = acc[r] + bias[c];
        }
    }
}

// ---------------------------------------------------------------------------
// Masked per-channel batch stats: two-stage deterministic reduction.
// ---------------------------------------------------------------------------
__global__ void stats_partial_kernel(const float* __restrict__ h, int stride, long E, int C,
                                     const int* __restrict__ valid, float* __restrict__ partial) {
    const int tid = threadIdx.x;
    float s = 0.f, ss = 0.f, cnt = 0.f;
    for (long r = blockIdx.x; r < E; r += gridDim.x) {
        const float m = valid ? (valid[r] ? 1.f : 0.f) : 1.f;
        if (tid < C) {
            float v = h[r * (long)stride + tid];
            s += v * m; ss += v * v * m;
        }
        if (tid == 0) cnt += m;
    }
    float* pb = partial + (long)blockIdx.x * (2 * C + 1);
    if (tid < C) { pb[tid] = s; pb[C + tid] = ss; }
    if (tid == 0) pb[2 * C] = cnt;
}

__global__ void stats_final_kernel(const float* __restrict__ partial, int G, int C, int KPAD,
                                   const float* __restrict__ gam, const float* __restrict__ bet,
                                   float* __restrict__ ssout) {
    const int c = threadIdx.x;
    __shared__ float scnt;
    if (c == 0) {
        float cnt = 0.f;
        for (int g = 0; g < G; ++g) cnt += partial[(long)g * (2 * C + 1) + 2 * C];
        scnt = fmaxf(cnt, 1.f);
    }
    __syncthreads();
    if (c < KPAD) {
        if (c < C) {
            float s = 0.f, ss = 0.f;
            for (int g = 0; g < G; ++g) {
                const float* pb = partial + (long)g * (2 * C + 1);
                s += pb[c]; ss += pb[C + c];
            }
            const float mean = s / scnt;
            float var = ss / scnt - mean * mean;
            var = fmaxf(var, 0.f);
            const float sc = gam[c] * rsqrtf(var + EPSBN);
            ssout[c] = sc;
            ssout[KPAD + c] = bet[c] - mean * sc;
        } else {
            ssout[c] = 0.f; ssout[KPAD + c] = 0.f;   // pad channels die in relu(0*x+0)
        }
    }
}

// ---------------------------------------------------------------------------
// Masked max aggregation over K neighbors; rows >= B*S zeroed (scatter).
// ---------------------------------------------------------------------------
__global__ void aggregate_kernel(const float* __restrict__ h, int hStride,
                                 const int* __restrict__ valid, float* __restrict__ xout,
                                 int cout, long BS, long total) {
    for (long i = (long)blockIdx.x * blockDim.x + threadIdx.x; i < total;
         i += (long)gridDim.x * blockDim.x) {
        const long row = i / cout; const int c = (int)(i % cout);
        if (row < BS) {
            float mx = -INFINITY;
            const long e0 = row * KKn;
            for (int k = 0; k < KKn; ++k)
                if (valid[e0 + k]) mx = fmaxf(mx, h[(e0 + k) * (long)hStride + c]);
            xout[i] = (mx == -INFINITY) ? 0.f : mx;
        } else {
            xout[i] = 0.f;
        }
    }
}

// Global max pool over P points per cloud (h stride 256)
__global__ void pool_kernel(const float* __restrict__ h, float* __restrict__ g) {
    const int b = blockIdx.x, c = threadIdx.x;
    float mx = -INFINITY;
    for (int p = 0; p < PP; ++p)
        mx = fmaxf(mx, h[((long)b * PP + p) * 256 + c]);
    g[b * 256 + c] = mx;
}

// Head MLP (256->128->64->2, ReLU, no BN) + log_softmax, 8 rows.
__global__ void head_kernel(const float* __restrict__ g,
                            const float* __restrict__ W0, const float* __restrict__ b0,
                            const float* __restrict__ W1, const float* __restrict__ b1,
                            const float* __restrict__ W2, const float* __restrict__ b2,
                            float* __restrict__ out) {
    __shared__ float l1[128], l2[64], l3[2];
    const int t = threadIdx.x;
    for (int b = 0; b < BB; ++b) {
        if (t < 128) {
            float a = b0[t];
            for (int i = 0; i < 256; ++i) a += g[b * 256 + i] * W0[i * 128 + t];
            l1[t] = fmaxf(a, 0.f);
        }
        __syncthreads();
        if (t < 64) {
            float a = b1[t];
            for (int i = 0; i < 128; ++i) a += l1[i] * W1[i * 64 + t];
            l2[t] = fmaxf(a, 0.f);
        }
        __syncthreads();
        if (t < 2) {
            float a = b2[t];
            for (int i = 0; i < 64; ++i) a += l2[i] * W2[i * 2 + t];
            l3[t] = a;
        }
        __syncthreads();
        if (t == 0) {
            const float m = fmaxf(l3[0], l3[1]);
            const float lse = m + logf(expf(l3[0] - m) + expf(l3[1] - m));
            out[b * 2 + 0] = l3[0] - lse;
            out[b * 2 + 1] = l3[1] - lse;
        }
        __syncthreads();
    }
}

// ---------------------------------------------------------------------------
extern "C" void kernel_launch(void* const* d_in, const int* in_sizes, int n_in,
                              void* d_out, int out_size, void* d_ws, size_t ws_size,
                              hipStream_t stream) {
    (void)in_sizes; (void)n_in; (void)out_size; (void)ws_size;
    // d_in order (jax dict flatten, keys sorted): pos, batch, head{W0,W1,W2,b0,b1,b2},
    //   sa1{W0,W1,W2,b0,b1,b2,be0,be1,g0,g1}, sa2{...}, sa3{...}
    const float* pos = (const float*)d_in[0];
    const float* hW0 = (const float*)d_in[2];  const float* hW1 = (const float*)d_in[3];
    const float* hW2 = (const float*)d_in[4];  const float* hb0 = (const float*)d_in[5];
    const float* hb1 = (const float*)d_in[6];  const float* hb2 = (const float*)d_in[7];
    const float* s1W[3]  = {(const float*)d_in[8],  (const float*)d_in[9],  (const float*)d_in[10]};
    const float* s1b[3]  = {(const float*)d_in[11], (const float*)d_in[12], (const float*)d_in[13]};
    const float* s1be[2] = {(const float*)d_in[14], (const float*)d_in[15]};
    const float* s1g[2]  = {(const float*)d_in[16], (const float*)d_in[17]};
    const float* s2W[3]  = {(const float*)d_in[18], (const float*)d_in[19], (const float*)d_in[20]};
    const float* s2b[3]  = {(const float*)d_in[21], (const float*)d_in[22], (const float*)d_in[23]};
    const float* s2be[2] = {(const float*)d_in[24], (const float*)d_in[25]};
    const float* s2g[2]  = {(const float*)d_in[26], (const float*)d_in[27]};
    const float* s3W[3]  = {(const float*)d_in[28], (const float*)d_in[29], (const float*)d_in[30]};
    const float* s3b[3]  = {(const float*)d_in[31], (const float*)d_in[32], (const float*)d_in[33]};
    const float* s3be[2] = {(const float*)d_in[34], (const float*)d_in[35]};
    const float* s3g[2]  = {(const float*)d_in[36], (const float*)d_in[37]};
    float* out = (float*)d_out;

    // bump allocator over d_ws
    char* base = (char*)d_ws; size_t off = 0;
    auto alloc = [&](size_t bytes) -> void* {
        off = (off + 255) & ~(size_t)255;
        void* p = base + off; off += bytes; return p;
    };
    int*   idx1 = (int*)alloc(sizeof(int) * BB * SS1);
    int*   idx2 = (int*)alloc(sizeof(int) * BB * SS2);
    int*   nbr1 = (int*)alloc(sizeof(int) * E1);
    int*   val1 = (int*)alloc(sizeof(int) * E1);
    int*   nbr2 = (int*)alloc(sizeof(int) * E2);
    int*   val2 = (int*)alloc(sizeof(int) * E2);
    float* bufA = (float*)alloc(sizeof(float) * E1 * 32);   // == E2*64 == max needed
    float* bufB = (float*)alloc(sizeof(float) * E1 * 32);
    float* x1   = (float*)alloc(sizeof(float) * (long)NNp * 32);
    float* x2   = (float*)alloc(sizeof(float) * (long)NNp * 64);
    float* gbuf = (float*)alloc(sizeof(float) * BB * 256);
    float* part = (float*)alloc(sizeof(float) * GSTAT * (2 * 128 + 1));
    float* ssb[6];
    for (int i = 0; i < 6; ++i) ssb[i] = (float*)alloc(sizeof(float) * 2 * 128);
    _Float16* wt[9];
    const int wcin[9]  = {3, 16, 16,  35, 32, 32,  67,  64, 128};
    const int wcout[9] = {16, 16, 32, 32, 32, 64,  64, 128, 256};
    const int wkp[9]   = {32, 32, 32, 64, 32, 32,  96,  64, 128};
    const float* wsrc[9] = {s1W[0], s1W[1], s1W[2], s2W[0], s2W[1], s2W[2], s3W[0], s3W[1], s3W[2]};
    for (int i = 0; i < 9; ++i) {
        wt[i] = (_Float16*)alloc(sizeof(_Float16) * (long)wcout[i] * wkp[i]);
        prep_w_kernel<<<64, 256, 0, stream>>>(wsrc[i], wcin[i], wcout[i], wkp[i], wt[i]);
    }

    // ---------------- SA1 ----------------
    fps_kernel<<<BB, 1024, 0, stream>>>(pos, idx1, SS1);
    knn_kernel<<<(BB * SS1) / 2, 64, 0, stream>>>(pos, idx1, SS1, 0.2f * 0.2f, nbr1, val1);
    gather1_kernel<<<4096, 256, 0, stream>>>(pos, nbr1, bufA);
    gemm_kernel<32, 16><<<dim3((unsigned)(E1 / 128), 1), 256, 0, stream>>>(bufA, wt[0], s1b[0], nullptr, bufB, 32);
    stats_partial_kernel<<<GSTAT, 256, 0, stream>>>(bufB, 32, E1, 16, val1, part);
    stats_final_kernel<<<1, 256, 0, stream>>>(part, GSTAT, 16, 32, s1g[0], s1be[0], ssb[0]);
    gemm_kernel<32, 16><<<dim3((unsigned)(E1 / 128), 1), 256, 0, stream>>>(bufB, wt[1], s1b[1], ssb[0], bufA, 32);
    stats_partial_kernel<<<GSTAT, 256, 0, stream>>>(bufA, 32, E1, 16, val1, part);
    stats_final_kernel<<<1, 256, 0, stream>>>(part, GSTAT, 16, 32, s1g[1], s1be[1], ssb[1]);
    gemm_kernel<32, 32><<<dim3((unsigned)(E1 / 128), 1), 256, 0, stream>>>(bufA, wt[2], s1b[2], ssb[1], bufB, 32);
    aggregate_kernel<<<4096, 256, 0, stream>>>(bufB, 32, val1, x1, 32, (long)BB * SS1, (long)NNp * 32);

    // ---------------- SA2 ----------------
    fps_kernel<<<BB, 1024, 0, stream>>>(pos, idx2, SS2);
    knn_kernel<<<(BB * SS2) / 2, 64, 0, stream>>>(pos, idx2, SS2, 0.4f * 0.4f, nbr2, val2);
    gather2_kernel<<<4096, 256, 0, stream>>>(pos, nbr2, x1, bufA);
    gemm_kernel<64, 32><<<dim3((unsigned)(E2 / 128), 1), 256, 0, stream>>>(bufA, wt[3], s2b[0], nullptr, bufB, 32);
    stats_partial_kernel<<<GSTAT, 256, 0, stream>>>(bufB, 32, E2, 32, val2, part);
    stats_final_kernel<<<1, 256, 0, stream>>>(part, GSTAT, 32, 32, s2g[0], s2be[0], ssb[2]);
    gemm_kernel<32, 32><<<dim3((unsigned)(E2 / 128), 1), 256, 0, stream>>>(bufB, wt[4], s2b[1], ssb[2], bufA, 32);
    stats_partial_kernel<<<GSTAT, 256, 0, stream>>>(bufA, 32, E2, 32, val2, part);
    stats_final_kernel<<<1, 256, 0, stream>>>(part, GSTAT, 32, 32, s2g[1], s2be[1], ssb[3]);
    gemm_kernel<32, 64><<<dim3((unsigned)(E2 / 128), 1), 256, 0, stream>>>(bufA, wt[5], s2b[2], ssb[3], bufB, 64);
    aggregate_kernel<<<4096, 256, 0, stream>>>(bufB, 64, val2, x2, 64, (long)BB * SS2, (long)NNp * 64);

    // ---------------- SA3 (per-node MLP) ----------------
    gather3_kernel<<<4096, 256, 0, stream>>>(x2, pos, bufA);
    gemm_kernel<96, 64><<<dim3(NNp / 128, 1), 256, 0, stream>>>(bufA, wt[6], s3b[0], nullptr, bufB, 64);
    stats_partial_kernel<<<GSTAT, 256, 0, stream>>>(bufB, 64, NNp, 64, nullptr, part);
    stats_final_kernel<<<1, 256, 0, stream>>>(part, GSTAT, 64, 64, s3g[0], s3be[0], ssb[4]);
    gemm_kernel<64, 64><<<dim3(NNp / 128, 2), 256, 0, stream>>>(bufB, wt[7], s3b[1], ssb[4], bufA, 128);
    stats_partial_kernel<<<GSTAT, 256, 0, stream>>>(bufA, 128, NNp, 128, nullptr, part);
    stats_final_kernel<<<1, 256, 0, stream>>>(part, GSTAT, 128, 128, s3g[1], s3be[1], ssb[5]);
    gemm_kernel<128, 64><<<dim3(NNp / 128, 4), 256, 0, stream>>>(bufA, wt[8], s3b[2], ssb[5], bufB, 256);

    // ---------------- pool + head + log_softmax ----------------
    pool_kernel<<<BB, 256, 0, stream>>>(bufB, gbuf);
    head_kernel<<<1, 256, 0, stream>>>(gbuf, hW0, hb0, hW1, hb1, hW2, hb2, out);
}
